// MaceEquivariantScorePredictionHead_12025908428970
// MI455X (gfx1250) — matrix operand order repaired and checked by reading, not compiled, not run on hardware
//
#include <hip/hip_runtime.h>
#include <hip/hip_bf16.h>

// MACE equivariant score head, algebraically collapsed for MI455X (gfx1250).
// Only the vector channel reaches the output; all layer transforms are linear
// per (node, xyz) row except a per-channel scale derived from a global Gram
// matrix G = X^T X  (X: 98304 x 128 rows of times*tp_norm*x_v).
// Heavy work = one WMMA f16->f32 Gram pass + one f32 output pass (mem-bound).

typedef __attribute__((ext_vector_type(16))) _Float16 v16h;
typedef __attribute__((ext_vector_type(8)))  _Float16 v8h;
typedef __attribute__((ext_vector_type(8)))  float    v8f;

#define N_NODES     32768
#define TPNORM      0.022097086912079608f   // 1/sqrt(16*128)
#define INV_SQRT_V  0.08838834764831845f    // 1/sqrt(128)
#define EPSV        1e-5f
#define RS          104                     // LDS row stride in halfs (96 rows + 8 pad)
#define NCHUNK      1024                    // 32768 nodes / 32 nodes per chunk

union V16 { v16h v; v8h h[2]; };

// ---------------------------------------------------------------------------
// prep: A1[v,o] = sum_u W_time[u] * W_tp1[u,v,o]; zero Gram accumulator; ones
// ---------------------------------------------------------------------------
__global__ __launch_bounds__(256) void prep_kernel(
    const float* __restrict__ Wtime, const float* __restrict__ Wtp1,
    float* __restrict__ A1, float* __restrict__ G, float* __restrict__ ones)
{
    int tid = blockIdx.x * 256 + threadIdx.x;     // 64 blocks -> 16384 threads
    int v = tid >> 7, o = tid & 127;
    float s = 0.f;
    #pragma unroll
    for (int u = 0; u < 16; ++u)
        s += Wtime[u] * Wtp1[u * 16384 + v * 128 + o];
    A1[tid] = s;
    G[tid]  = 0.f;
    if (tid < 128) ones[tid] = 1.0f;
}

// ---------------------------------------------------------------------------
// Gram: G += X^T X using v_wmma_f32_16x16x32_f16.
// Each block stages 32 nodes (96 X-rows) into LDS (f16, column-major,
// padded stride); each of the 8 waves owns one 16-row block of G and
// accumulates all 8 column blocks across chunks; flushed via atomics.
// All 8 B fragments are preloaded per K-chunk so the LDS loads batch up
// ahead of the back-to-back WMMA sequence (wait-ladder instead of stalls).
// ---------------------------------------------------------------------------
__global__ __launch_bounds__(256) void gram_kernel(
    const float* __restrict__ feat, const float* __restrict__ times,
    float* __restrict__ G)
{
    __shared__ __align__(16) _Float16 ldsX[128 * RS];
    __shared__ float ldsT[32];

    const int tid  = threadIdx.x;
    const int lane = tid & 31;
    const int wave = tid >> 5;          // 0..7 -> G row-block
    const int lo   = lane & 15;
    const int hi   = lane >> 4;         // 0/1

    v8f acc[8];
    #pragma unroll
    for (int nb = 0; nb < 8; ++nb)
        #pragma unroll
        for (int j = 0; j < 8; ++j) acc[nb][j] = 0.f;

    for (int chunk = blockIdx.x; chunk < NCHUNK; chunk += gridDim.x) {
        __syncthreads();                           // LDS reuse fence
        const int nb0 = chunk * 32;
        if (tid < 32) ldsT[tid] = times[nb0 + tid] * TPNORM;
        __syncthreads();

        const bool more = (chunk + gridDim.x) < NCHUNK;   // block-uniform

        // Stage 32 nodes x 384 f32 -> f16 LDS, column-major [v][row], row=3*ln+m
        #pragma unroll
        for (int j = 0; j < 12; ++j) {
            int idx = tid + 256 * j;               // 0..3071 float4 slots
            int ln  = idx / 96;
            int q   = idx - ln * 96;               // float4 index within 384
            const float* gp = feat + (size_t)(nb0 + ln) * 512 + 128 + 4 * q;
            const float4 f4 = *(const float4*)gp;
            if (more)                              // warm L2/WGP$ for next chunk
                __builtin_prefetch(gp + (size_t)gridDim.x * 32 * 512, 0, 3);
            float s = ldsT[ln];
            float vals[4] = { f4.x * s, f4.y * s, f4.z * s, f4.w * s };
            int f = 4 * q;
            #pragma unroll
            for (int e = 0; e < 4; ++e) {
                int ff = f + e;
                int vc = ff / 3;                   // channel (LDS column)
                int m  = ff - 3 * vc;              // xyz component
                ldsX[vc * RS + ln * 3 + m] = (_Float16)vals[e];
            }
        }
        __syncthreads();

        // 3 K-chunks of 32 rows; A = X^T tile (this wave's row block), B = X tile
        #pragma unroll
        for (int k0 = 0; k0 < 96; k0 += 32) {
            // A 16x32 f16 layout: lane<16: K0..7 | K16..23 ; lane>=16: K8..15 | K24..31
            const _Float16* pa = &ldsX[(16 * wave + lo) * RS + k0 + hi * 8];
            V16 a;
            a.h[0] = *(const v8h*)pa;
            a.h[1] = *(const v8h*)(pa + 16);

            // Preload all 8 B fragments (32x16 f16: lane<16 K0..7 | K16..23; lane>=16 K8..15 | K24..31)
            V16 bf[8];
            #pragma unroll
            for (int nb = 0; nb < 8; ++nb) {
                const _Float16* pb = &ldsX[(16 * nb + lo) * RS + k0 + hi * 8];
                bf[nb].h[0] = *(const v8h*)pb;
                bf[nb].h[1] = *(const v8h*)(pb + 16);
            }
            #pragma unroll
            for (int nb = 0; nb < 8; ++nb)
                acc[nb] = __builtin_amdgcn_wmma_f32_16x16x32_f16(
                    false, a.v, false, bf[nb].v, (short)0, acc[nb], false, false);
        }
    }

    // Flush partial Gram: C/D layout -> row = 16*wave + j + 8*hi, col = 16*nb + lo
    #pragma unroll
    for (int nb = 0; nb < 8; ++nb)
        #pragma unroll
        for (int j = 0; j < 8; ++j) {
            int row = 16 * wave + j + 8 * hi;
            int col = 16 * nb + lo;
            atomicAdd(&G[row * 128 + col], acc[nb][j]);
        }
}

// ---------------------------------------------------------------------------
// Bout[v,o] = (1/sqrt(128)) * sum_u Bin[v,u] * scale[u] * Wv[u,o]
// ---------------------------------------------------------------------------
__global__ __launch_bounds__(256) void bmat_kernel(
    const float* __restrict__ Bin, const float* __restrict__ scale,
    const float* __restrict__ Wv, float* __restrict__ Bout)
{
    int tid = blockIdx.x * 256 + threadIdx.x;     // 64 blocks -> 16384
    int v = tid >> 7, o = tid & 127;
    float s = 0.f;
    #pragma unroll 8
    for (int u = 0; u < 128; ++u)
        s += Bin[v * 128 + u] * scale[u] * Wv[u * 128 + o];
    Bout[tid] = s * INV_SQRT_V;
}

// ---------------------------------------------------------------------------
// scale[o] = gamma[o] * rsqrt( (B^T G B)_oo / (3N) + eps )
// one block per output channel o
// ---------------------------------------------------------------------------
__global__ __launch_bounds__(128) void fnscale_kernel(
    const float* __restrict__ B, const float* __restrict__ G,
    const float* __restrict__ gamma, float* __restrict__ scale_out)
{
    const int o = blockIdx.x;
    const int v = threadIdx.x;
    float t = 0.f;
    #pragma unroll 8
    for (int u = 0; u < 128; ++u)
        t += G[v * 128 + u] * B[u * 128 + o];
    __shared__ float red[128];
    red[v] = B[v * 128 + o] * t;
    __syncthreads();
    #pragma unroll
    for (int s = 64; s > 0; s >>= 1) {
        if (v < s) red[v] += red[v + s];
        __syncthreads();
    }
    if (v == 0)
        scale_out[o] = gamma[o] * rsqrtf(red[0] * (1.0f / (3.0f * N_NODES)) + EPSV);
}

// ---------------------------------------------------------------------------
// w[v] = (1/sqrt(128)) * sum_u B2[v,u] * scale2[u] * Wr[u]
// ---------------------------------------------------------------------------
__global__ __launch_bounds__(128) void wvec_kernel(
    const float* __restrict__ B2, const float* __restrict__ scale2,
    const float* __restrict__ Wr, float* __restrict__ w)
{
    int v = threadIdx.x;
    float s = 0.f;
    #pragma unroll 8
    for (int u = 0; u < 128; ++u)
        s += B2[v * 128 + u] * scale2[u] * Wr[u];
    w[v] = s * INV_SQRT_V;
}

// ---------------------------------------------------------------------------
// scores[n,m] = times[n]*tp_norm * sum_v x_v[n,v,m] * w[v]   (one wave / node)
// lane reads 12 consecutive floats (4 channels x 3 components), f32 exact
// ---------------------------------------------------------------------------
__global__ __launch_bounds__(256) void scores_kernel(
    const float* __restrict__ feat, const float* __restrict__ times,
    const float* __restrict__ w, float* __restrict__ out)
{
    const int lane = threadIdx.x & 31;
    const int wave = threadIdx.x >> 5;
    const int n = blockIdx.x * 8 + wave;          // 4096 blocks * 8 waves = 32768

    const float* p = feat + (size_t)n * 512 + 128 + lane * 12;
    float4 f0 = *(const float4*)(p);
    float4 f1 = *(const float4*)(p + 4);
    float4 f2 = *(const float4*)(p + 8);
    float4 wv = *(const float4*)(w + lane * 4);   // channels 4*lane .. 4*lane+3

    // f = 12*lane + e : channel = 4*lane + e/3, component = e%3
    float pm0 = f0.x * wv.x + f0.w * wv.y + f1.z * wv.z + f2.y * wv.w;  // m=0
    float pm1 = f0.y * wv.x + f1.x * wv.y + f1.w * wv.z + f2.z * wv.w;  // m=1
    float pm2 = f0.z * wv.x + f1.y * wv.y + f2.x * wv.z + f2.w * wv.w;  // m=2

    #pragma unroll
    for (int off = 16; off > 0; off >>= 1) {
        pm0 += __shfl_down(pm0, off, 32);
        pm1 += __shfl_down(pm1, off, 32);
        pm2 += __shfl_down(pm2, off, 32);
    }
    if (lane == 0) {
        float sc = times[n] * TPNORM;
        out[n * 3 + 0] = pm0 * sc;
        out[n * 3 + 1] = pm1 * sc;
        out[n * 3 + 2] = pm2 * sc;
    }
}

// ---------------------------------------------------------------------------
extern "C" void kernel_launch(void* const* d_in, const int* in_sizes, int n_in,
                              void* d_out, int out_size, void* d_ws, size_t ws_size,
                              hipStream_t stream)
{
    const float* feat    = (const float*)d_in[0];   // (N, 512)
    const float* times   = (const float*)d_in[1];   // (N, 1)
    const float* Wtime   = (const float*)d_in[2];   // (1, 16)
    // d_in[3] = W_tp0, d_in[5] = Ws, d_in[7] = gamma_s, d_in[8] = beta_s: dead
    const float* Wtp1    = (const float*)d_in[4];   // (16, 128, 128)
    const float* Wv      = (const float*)d_in[6];   // (3, 128, 128)
    const float* gamma_v = (const float*)d_in[9];   // (3, 128)
    const float* Wr      = (const float*)d_in[10];  // (128, 1)
    float* out = (float*)d_out;                     // (N, 3)

    float* ws   = (float*)d_ws;
    float* buf0 = ws;                 // 16384
    float* buf1 = ws + 16384;         // 16384
    float* G    = ws + 32768;         // 16384
    float* ones = ws + 49152;         // 128
    float* sc0  = ws + 49280;         // 128
    float* sc1  = ws + 49408;         // 128
    float* sc2  = ws + 49536;         // 128
    float* w    = ws + 49664;         // 128

    prep_kernel<<<64, 256, 0, stream>>>(Wtime, Wtp1, buf0, G, ones);
    gram_kernel<<<512, 256, 0, stream>>>(feat, times, G);   // 2 blocks/WGP overlap

    bmat_kernel<<<64, 256, 0, stream>>>(buf0, ones, Wv,          buf1);
    fnscale_kernel<<<128, 128, 0, stream>>>(buf1, G, gamma_v,       sc0);
    bmat_kernel<<<64, 256, 0, stream>>>(buf1, sc0,  Wv + 16384,  buf0);
    fnscale_kernel<<<128, 128, 0, stream>>>(buf0, G, gamma_v + 128, sc1);
    bmat_kernel<<<64, 256, 0, stream>>>(buf0, sc1,  Wv + 32768,  buf1);
    fnscale_kernel<<<128, 128, 0, stream>>>(buf1, G, gamma_v + 256, sc2);

    wvec_kernel<<<1, 128, 0, stream>>>(buf1, sc2, Wr, w);
    scores_kernel<<<4096, 256, 0, stream>>>(feat, times, w, out);
}
